// TransformerLanguageModel_4355096838320
// MI455X (gfx1250) — compile-verified, matrix operand order checked by reading, not run on hardware
//
#include <hip/hip_runtime.h>
#include <hip/hip_bf16.h>

// ---------------------------------------------------------------------------
// CDNA5 (gfx1250) transformer forward pass using v_wmma_f32_16x16x32_f16
// plus Tensor-Data-Mover (tensor_load_to_lds) double-buffered GEMM staging.
// wave32; WMMA 16x16x32 f16->f32.  A-fragment layout (16-bit, 16x32 MxK):
//   lane<16 : row M=lane,    K = {0..7} in v0..3, {16..23} in v4..7
//   lane>=16: row M=lane-16, K = {8..15},{24..31}
// B-fragment assumed symmetric with N in place of M (B supplied as Bt[N][K]).
// C/D layout: VGPR v, lanes0-15 -> (M=v, N=lane), lanes16-31 -> (M=v+8).
// ---------------------------------------------------------------------------

typedef _Float16 v16h __attribute__((ext_vector_type(16)));
typedef _Float16 v8h  __attribute__((ext_vector_type(8)));
typedef float    v8f  __attribute__((ext_vector_type(8)));
typedef unsigned int u32x4 __attribute__((ext_vector_type(4)));
typedef int          i32x4 __attribute__((ext_vector_type(4)));
typedef int          i32x8 __attribute__((ext_vector_type(8)));

__device__ static inline v16h ld_frag16x32(const _Float16* p, int laneHi) {
  // p points at (row, k0); loads the lane's 2x8 halves for a 16x32 fragment.
  const v8h lo = *(const v8h*)(p + (laneHi << 3));
  const v8h hi = *(const v8h*)(p + 16 + (laneHi << 3));
  v16h a;
#pragma unroll
  for (int i = 0; i < 8; ++i) { a[i] = lo[i]; a[i + 8] = hi[i]; }
  return a;
}

__device__ static inline v8f wmma_f16(v16h a, v16h b, v8f c) {
  return __builtin_amdgcn_wmma_f32_16x16x32_f16(
      /*neg_a=*/false, a, /*neg_b=*/false, b,
      /*c_mod=*/(short)0, c, /*reuse_a=*/false, /*reuse_b=*/false);
}

// ---------------------------------------------------------------------------
// TDM: DMA a 2D f16 tile (tile_h rows x 32 cols, row stride K elems) from
// global memory into LDS at byte offset lds_off (rows stored contiguously,
// 32 elems each).  Descriptor per CDNA5 ISA section 8 (D# groups 0/1).
// ---------------------------------------------------------------------------
__device__ static inline void tdm_load_tile32(const _Float16* gptr,
                                              unsigned lds_off,
                                              int K, int tile_h) {
  const unsigned long long ga = (unsigned long long)(uintptr_t)gptr;
  u32x4 g0;
  g0[0] = 1u;                                         // count=1, user D#
  g0[1] = lds_off;                                    // LDS byte address
  g0[2] = (unsigned)(ga & 0xffffffffu);               // global_addr[31:0]
  g0[3] = (unsigned)((ga >> 32) & 0x01ffffffu) | (2u << 30);  // [56:32]|type=2
  i32x8 g1;
  g1[0] = 0x00010000;                                 // data_size=1 (2 bytes)
  g1[1] = (int)((unsigned)(K & 0xffff) << 16);        // tensor_dim0[15:0]
  g1[2] = (int)(((unsigned)(K >> 16) & 0xffffu) |     // tensor_dim0[31:16]
                ((unsigned)(tile_h & 0xffff) << 16)); // tensor_dim1[15:0]
  g1[3] = (int)((32u) << 16);                         // tile_dim0=32
  g1[4] = (int)(tile_h & 0xffff);                     // tile_dim1=tile_h
  g1[5] = K;                                          // tensor_dim0_stride lo
  g1[6] = 0;
  g1[7] = 0;
  const i32x4 z4 = {0, 0, 0, 0};
#if __clang_major__ >= 23
  const i32x8 z8 = {};
  __builtin_amdgcn_tensor_load_to_lds(g0, g1, z4, z4, z8, 0);
#else
  __builtin_amdgcn_tensor_load_to_lds(g0, g1, z4, z4, 0);
#endif
}

// ---------------------------------------------------------------------------
// Generic GEMM: C[M,N] = A[M,K](f16) * Bt[N,K](f16) (+bias)(+resid)(ReLU)
// FLAGS: 1=bias, 2=residual(read+write f32 resid), 4=relu, 8=f16 out, 16=f32 out
// block = 128 threads (4 waves); block tile 64 rows x 64 cols.
// A/B K-chunks (64x32 f16 each) are TDM-DMA'd into double-buffered LDS;
// wave0 issues the DMA, everyone consumes via ds_load_b128 fragments.
// ---------------------------------------------------------------------------
template <int FLAGS>
__global__ __launch_bounds__(128) void gemm_wmma(
    const _Float16* __restrict__ A, const _Float16* __restrict__ Bt,
    const float* __restrict__ bias, float* __restrict__ resid,
    float* __restrict__ out32, _Float16* __restrict__ out16,
    int M, int N, int K) {
  __shared__ _Float16 sA[2][64 * 32];
  __shared__ _Float16 sB[2][64 * 32];

  const int lane   = threadIdx.x & 31;
  const int wave   = threadIdx.x >> 5;
  const int laneLo = lane & 15;
  const int laneHi = lane >> 4;
  const int m0 = blockIdx.y * 64 + wave * 16;
  const int n0 = blockIdx.x * 64;
  const bool issuer = (threadIdx.x < 32);

  const _Float16* Ablk = A + (size_t)(blockIdx.y * 64) * K;
  const _Float16* Bblk = Bt + (size_t)n0 * K;
  const unsigned offA[2] = {(unsigned)(uintptr_t)(void*)&sA[0][0],
                            (unsigned)(uintptr_t)(void*)&sA[1][0]};
  const unsigned offB[2] = {(unsigned)(uintptr_t)(void*)&sB[0][0],
                            (unsigned)(uintptr_t)(void*)&sB[1][0]};

  const int nk = K >> 5;
  if (issuer) {
    tdm_load_tile32(Ablk, offA[0], K, 64);
    tdm_load_tile32(Bblk, offB[0], K, 64);
  }

  v8f acc[4] = {};
  for (int i = 0; i < nk; ++i) {
    const int buf = i & 1;
    if (issuer) __builtin_amdgcn_s_wait_tensorcnt(0);
    __syncthreads();                       // tile i visible to all waves
    if (issuer && (i + 1) < nk) {          // overlap next DMA with compute
      tdm_load_tile32(Ablk + (size_t)(i + 1) * 32, offA[buf ^ 1], K, 64);
      tdm_load_tile32(Bblk + (size_t)(i + 1) * 32, offB[buf ^ 1], K, 64);
    }
    const v16h a = ld_frag16x32(&sA[buf][(wave * 16 + laneLo) * 32], laneHi);
#pragma unroll
    for (int nb = 0; nb < 4; ++nb) {
      const v16h b = ld_frag16x32(&sB[buf][(nb * 16 + laneLo) * 32], laneHi);
      acc[nb] = wmma_f16(a, b, acc[nb]);
    }
    __syncthreads();                       // done reading buf before reuse
  }

#pragma unroll
  for (int nb = 0; nb < 4; ++nb) {
    const int col = n0 + nb * 16 + laneLo;
    float bv = 0.f;
    if (FLAGS & 1) bv = bias[col];
#pragma unroll
    for (int i = 0; i < 8; ++i) {
      const int row = m0 + i + 8 * laneHi;
      const size_t off = (size_t)row * N + col;
      float v = acc[nb][i] + bv;
      if (FLAGS & 2) v += resid[off];
      if (FLAGS & 4) v = v > 0.f ? v : 0.f;
      if (FLAGS & 2) resid[off] = v;
      if (FLAGS & 16) out32[off] = v;
      if (FLAGS & 8) out16[off] = (_Float16)v;
    }
  }
}

// ---------------------------------------------------------------------------
// Flash attention: 1 wave per (b, h, 16-row query tile).  Causal softmax,
// online max/sum; P tile goes through LDS to convert C-layout -> A-layout.
// q,k : f16 [B,T,D] (head slice at h*HS);  vT : f16 [B,H,HS,T];  o : f16 [B,T,D]
// ---------------------------------------------------------------------------
__global__ __launch_bounds__(32) void attention_kernel(
    const _Float16* __restrict__ q, const _Float16* __restrict__ k,
    const _Float16* __restrict__ vT, _Float16* __restrict__ o,
    int T, int D, int H, int HS, float scale) {
  __shared__ _Float16 lds[16 * 32];

  const int lane   = threadIdx.x & 31;
  const int laneLo = lane & 15;
  const int laneHi = lane >> 4;
  const int nqt = T / 16;
  const int qi = blockIdx.x % nqt;
  const int h  = (blockIdx.x / nqt) % H;
  const int b  = blockIdx.x / (nqt * H);
  const int t0 = qi * 16;

  const _Float16* qbase = q + (size_t)(b * T + t0 + laneLo) * D + h * HS;
  const v16h aq0 = ld_frag16x32(qbase, laneHi);
  const v16h aq1 = ld_frag16x32(qbase + 32, laneHi);
  const _Float16* vTp = vT + ((size_t)(b * H + h)) * HS * T;

  v8f acc[4] = {};
  float m_i[8], l_i[8];
#pragma unroll
  for (int i = 0; i < 8; ++i) { m_i[i] = -1e30f; l_i[i] = 0.f; }

  for (int j0 = 0; j0 <= t0 + 15; j0 += 32) {
    // --- S = Q K^T (two 16x16 tiles covering 32 keys) ---
    const _Float16* kb0 = k + (size_t)(b * T + j0 + laneLo) * D + h * HS;
    const _Float16* kb1 = kb0 + (size_t)16 * D;
    v8f s0 = {}, s1 = {};
    s0 = wmma_f16(aq0, ld_frag16x32(kb0, laneHi), s0);
    s0 = wmma_f16(aq1, ld_frag16x32(kb0 + 32, laneHi), s0);
    s1 = wmma_f16(aq0, ld_frag16x32(kb1, laneHi), s1);
    s1 = wmma_f16(aq1, ld_frag16x32(kb1 + 32, laneHi), s1);

    const int col0 = j0 + laneLo;
    const int col1 = j0 + 16 + laneLo;
#pragma unroll
    for (int i = 0; i < 8; ++i) {
      const int row = t0 + i + 8 * laneHi;
      const float f0 = (col0 <= row) ? s0[i] * scale : -1e30f;
      const float f1 = (col1 <= row) ? s1[i] * scale : -1e30f;
      float rmax = fmaxf(f0, f1);
#pragma unroll
      for (int d = 1; d < 16; d <<= 1) rmax = fmaxf(rmax, __shfl_xor(rmax, d, 32));
      const float mn = fmaxf(m_i[i], rmax);
      const float alpha = __expf(m_i[i] - mn);
      m_i[i] = mn;
      const float p0 = __expf(f0 - mn);
      const float p1 = __expf(f1 - mn);
      float rs = p0 + p1;
#pragma unroll
      for (int d = 1; d < 16; d <<= 1) rs += __shfl_xor(rs, d, 32);
      l_i[i] = l_i[i] * alpha + rs;
#pragma unroll
      for (int nb = 0; nb < 4; ++nb) acc[nb][i] *= alpha;
      lds[(i + 8 * laneHi) * 32 + laneLo]      = (_Float16)p0;
      lds[(i + 8 * laneHi) * 32 + 16 + laneLo] = (_Float16)p1;
    }

    // Reload P as an A-fragment (16x32).  LDS ops are in-order per wave.
    const v16h pf = ld_frag16x32(&lds[laneLo * 32], laneHi);

    // --- O += P V ---
#pragma unroll
    for (int nb = 0; nb < 4; ++nb) {
      const _Float16* vrow = vTp + (size_t)(nb * 16 + laneLo) * T + j0;
      acc[nb] = wmma_f16(pf, ld_frag16x32(vrow, laneHi), acc[nb]);
    }
  }

#pragma unroll
  for (int nb = 0; nb < 4; ++nb) {
#pragma unroll
    for (int i = 0; i < 8; ++i) {
      const int t = t0 + i + 8 * laneHi;
      o[(size_t)(b * T + t) * D + h * HS + nb * 16 + laneLo] =
          (_Float16)(acc[nb][i] / l_i[i]);
    }
  }
}

// ---------------------------------------------------------------------------
// LayerNorm: 1 wave per row (D=512), f32 in -> f16 out.
// ---------------------------------------------------------------------------
__global__ __launch_bounds__(32) void layernorm_kernel(
    const float* __restrict__ x, const float* __restrict__ g,
    const float* __restrict__ b, _Float16* __restrict__ out) {
  const int row = blockIdx.x;
  const int lane = threadIdx.x;
  const float* xr = x + (size_t)row * 512;
  float4 v[4];
  float s = 0.f, ss = 0.f;
#pragma unroll
  for (int i = 0; i < 4; ++i) {
    v[i] = *(const float4*)(xr + i * 128 + lane * 4);
    s  += v[i].x + v[i].y + v[i].z + v[i].w;
    ss += v[i].x * v[i].x + v[i].y * v[i].y + v[i].z * v[i].z + v[i].w * v[i].w;
  }
#pragma unroll
  for (int d = 1; d < 32; d <<= 1) { s += __shfl_xor(s, d, 32); ss += __shfl_xor(ss, d, 32); }
  const float mu = s * (1.f / 512.f);
  const float var = ss * (1.f / 512.f) - mu * mu;
  const float rstd = rsqrtf(var + 1e-5f);
  _Float16* orow = out + (size_t)row * 512;
#pragma unroll
  for (int i = 0; i < 4; ++i) {
    const int c = i * 128 + lane * 4;
    const float4 gv = *(const float4*)(g + c);
    const float4 bv = *(const float4*)(b + c);
    orow[c + 0] = (_Float16)((v[i].x - mu) * rstd * gv.x + bv.x);
    orow[c + 1] = (_Float16)((v[i].y - mu) * rstd * gv.y + bv.y);
    orow[c + 2] = (_Float16)((v[i].z - mu) * rstd * gv.z + bv.z);
    orow[c + 3] = (_Float16)((v[i].w - mu) * rstd * gv.w + bv.w);
  }
}

// ---------------------------------------------------------------------------
// Embedding: x[b,t,:] = tok_emb[idx[b,t],:] + pos_emb[t,:]
// ---------------------------------------------------------------------------
__global__ void embed_kernel(const int* __restrict__ idx,
                             const float* __restrict__ tok,
                             const float* __restrict__ pos,
                             float* __restrict__ x, int T, int D) {
  const int row = blockIdx.x;
  const int t = row % T;
  const int token = idx[row];
  for (int c = threadIdx.x; c < D; c += blockDim.x)
    x[(size_t)row * D + c] = tok[(size_t)token * D + c] + pos[(size_t)t * D + c];
}

// ---------------------------------------------------------------------------
// Convert+transpose weights: W f32 [Lb][R][C] -> Wt f16 [Lb][C][R]
// ---------------------------------------------------------------------------
__global__ void cvt_transpose(const float* __restrict__ W,
                              _Float16* __restrict__ Wt,
                              int R, int C, long total) {
  const long t = (long)blockIdx.x * 256 + threadIdx.x;
  if (t >= total) return;
  const long mc = (long)R * C;
  const long mat = t / mc;
  const long rem = t - mat * mc;
  const int r = (int)(rem / C);
  const int c = (int)(rem - (long)r * C);
  Wt[mat * mc + (long)c * R + r] = (_Float16)W[t];
}

// ---------------------------------------------------------------------------
// vT[b,h,d,t] = v[b,t,h*HS+d]   (f16 -> f16 transpose for P@V fragments)
// ---------------------------------------------------------------------------
__global__ void make_vT(const _Float16* __restrict__ v,
                        _Float16* __restrict__ vT,
                        int Bn, int T, int H, int HS) {
  const long i = (long)blockIdx.x * 256 + threadIdx.x;
  const long total = (long)Bn * T * H * HS;
  if (i >= total) return;
  const int d = (int)(i % HS);
  long r = i / HS;
  const int h = (int)(r % H);
  r /= H;
  const int t = (int)(r % T);
  const int b = (int)(r / T);
  vT[(((long)b * H + h) * HS + d) * T + t] = v[i];
}

// ---------------------------------------------------------------------------
// Host orchestration
// ---------------------------------------------------------------------------
extern "C" void kernel_launch(void* const* d_in, const int* in_sizes, int n_in,
                              void* d_out, int out_size, void* d_ws, size_t ws_size,
                              hipStream_t stream) {
  (void)in_sizes; (void)n_in; (void)out_size; (void)ws_size;

  constexpr int Bn = 2, T = 2048, D = 512, H = 8, HS = 64, L = 4, V = 32000;
  constexpr int M = Bn * T;     // 4096
  constexpr int D4 = 4 * D;     // 2048

  const int*   idx    = (const int*)d_in[0];
  const float* tok    = (const float*)d_in[1];
  const float* pos    = (const float*)d_in[2];
  const float* ln1_g  = (const float*)d_in[3];
  const float* ln1_b  = (const float*)d_in[4];
  const float* wq     = (const float*)d_in[5];
  const float* wk     = (const float*)d_in[6];
  const float* wv     = (const float*)d_in[7];
  const float* wo     = (const float*)d_in[8];
  const float* bo     = (const float*)d_in[9];
  const float* ln2_g  = (const float*)d_in[10];
  const float* ln2_b  = (const float*)d_in[11];
  const float* w1     = (const float*)d_in[12];
  const float* b1     = (const float*)d_in[13];
  const float* w2     = (const float*)d_in[14];
  const float* b2     = (const float*)d_in[15];
  const float* lnf_g  = (const float*)d_in[16];
  const float* lnf_b  = (const float*)d_in[17];
  const float* head_w = (const float*)d_in[18];
  const float* head_b = (const float*)d_in[19];

  char* ws = (char*)d_ws;
  size_t off = 0;
  auto alloc = [&](size_t bytes) -> char* {
    char* p = ws + off;
    off = (off + bytes + 255) & ~(size_t)255;
    return p;
  };

  float*     x    = (float*)alloc((size_t)M * D * 4);
  _Float16*  h16  = (_Float16*)alloc((size_t)M * D * 2);
  _Float16*  q16  = (_Float16*)alloc((size_t)M * D * 2);
  _Float16*  k16  = (_Float16*)alloc((size_t)M * D * 2);
  _Float16*  v16  = (_Float16*)alloc((size_t)M * D * 2);
  _Float16*  vTb  = (_Float16*)alloc((size_t)Bn * H * HS * T * 2);
  _Float16*  att  = (_Float16*)alloc((size_t)M * D * 2);
  _Float16*  h1   = (_Float16*)alloc((size_t)M * D4 * 2);
  _Float16*  tq   = (_Float16*)alloc((size_t)L * D * D * 2);
  _Float16*  tk   = (_Float16*)alloc((size_t)L * D * D * 2);
  _Float16*  tv   = (_Float16*)alloc((size_t)L * D * D * 2);
  _Float16*  to   = (_Float16*)alloc((size_t)L * D * D * 2);
  _Float16*  t1   = (_Float16*)alloc((size_t)L * D * D4 * 2);
  _Float16*  t2   = (_Float16*)alloc((size_t)L * D4 * D * 2);
  _Float16*  th   = (_Float16*)alloc((size_t)D * V * 2);

  auto cvt = [&](const float* W, _Float16* Wt, int R, int C, long lb) {
    const long total = lb * (long)R * C;
    cvt_transpose<<<dim3((unsigned)((total + 255) / 256)), 256, 0, stream>>>(W, Wt, R, C, total);
  };
  cvt(wq, tq, D, D, L);
  cvt(wk, tk, D, D, L);
  cvt(wv, tv, D, D, L);
  cvt(wo, to, D, D, L);
  cvt(w1, t1, D, D4, L);
  cvt(w2, t2, D4, D, L);
  cvt(head_w, th, D, V, 1);

  embed_kernel<<<M, 128, 0, stream>>>(idx, tok, pos, x, T, D);

  const dim3 gemmBlk(128);
  const dim3 gDD(D / 64, M / 64);    // N=512
  const dim3 gD4(D4 / 64, M / 64);   // N=2048
  const dim3 gV(V / 64, M / 64);     // N=32000

  for (int l = 0; l < L; ++l) {
    layernorm_kernel<<<M, 32, 0, stream>>>(x, ln1_g + (size_t)l * D, ln1_b + (size_t)l * D, h16);

    gemm_wmma<8><<<gDD, gemmBlk, 0, stream>>>(h16, tq + (size_t)l * D * D, nullptr, nullptr, nullptr, q16, M, D, D);
    gemm_wmma<8><<<gDD, gemmBlk, 0, stream>>>(h16, tk + (size_t)l * D * D, nullptr, nullptr, nullptr, k16, M, D, D);
    gemm_wmma<8><<<gDD, gemmBlk, 0, stream>>>(h16, tv + (size_t)l * D * D, nullptr, nullptr, nullptr, v16, M, D, D);

    make_vT<<<dim3((unsigned)(((long)M * D + 255) / 256)), 256, 0, stream>>>(v16, vTb, Bn, T, H, HS);

    attention_kernel<<<Bn * H * (T / 16), 32, 0, stream>>>(q16, k16, vTb, att, T, D, H, HS, 0.125f);

    // x = x + att @ Wo + bo
    gemm_wmma<3><<<gDD, gemmBlk, 0, stream>>>(att, to + (size_t)l * D * D, bo + (size_t)l * D, x, nullptr, nullptr, M, D, D);

    layernorm_kernel<<<M, 32, 0, stream>>>(x, ln2_g + (size_t)l * D, ln2_b + (size_t)l * D, h16);

    // h1 = relu(h @ W1 + b1)   (f16 out)
    gemm_wmma<13><<<gD4, gemmBlk, 0, stream>>>(h16, t1 + (size_t)l * D * D4, b1 + (size_t)l * D4, nullptr, nullptr, h1, M, D4, D);

    // x = x + h1 @ W2 + b2
    gemm_wmma<3><<<gDD, gemmBlk, 0, stream>>>(h1, t2 + (size_t)l * D4 * D, b2 + (size_t)l * D, x, nullptr, nullptr, M, D, D4);
  }

  layernorm_kernel<<<M, 32, 0, stream>>>(x, lnf_g, lnf_b, h16);

  // logits = ln_f(x) @ head_w + head_b  -> d_out (f32)
  gemm_wmma<17><<<gV, gemmBlk, 0, stream>>>(h16, th, head_b, nullptr, (float*)d_out, nullptr, M, V, D);
}
